// UtterancePolicy_21560735826284
// MI455X (gfx1250) — compile-verified, matrix operand order checked by reading, not run on hardware
//
#include <hip/hip_runtime.h>

// ---------------- problem constants ----------------
#define EDIM 100     // hidden size
#define TTOK 10      // vocabulary / tokens
#define LSTEP 6      // decode steps
#define KP   160     // padded K: 0..99 h, 100..127 zero, 128..143 one-hot(tok), 144 const-1
#define NGRP 7       // ceil(100/16) hidden-unit groups
#define NTILE (NGRP*4)               // 28 gate tiles of 16 columns, order (g, {i,f,g,o})
#define WELEMS ((NTILE*16 + 16) * KP)        // gate tiles + w_out tile, f16 elements
#define WBYTES (WELEMS * 2)                  // 148480 bytes
#define HBYTES (8 * 16 * KP * 2)             // 40960 bytes  (8 waves x 16 rows x KP f16)
#define LDS_TOTAL (WBYTES + HBYTES + 8*16*4) // + tok array

typedef __attribute__((ext_vector_type(16))) _Float16 v16h;
typedef __attribute__((ext_vector_type(8)))  float    v8f;

__device__ inline float rng01(unsigned x) {
    x ^= x >> 16; x *= 0x7feb352dU; x ^= x >> 15; x *= 0x846ca68bU; x ^= x >> 16;
    return (float)(x >> 8) * (1.0f / 16777216.0f);
}

__device__ inline float lane_bperm(float v, int srcLane) {
    return __int_as_float(__builtin_amdgcn_ds_bpermute((srcLane & 31) << 2,
                                                       __float_as_int(v)));
}

// ---------------- prep: pack weights into WMMA-B layout (global scratch) ----------
// Wpack[tile][n][k] f16, k contiguous: a B fragment for (tile,kt) is a contiguous
// 32B per-lane load (lane = N | K-half).
//   tile t = g*4+q (group g, torch gate order i,f,g,o); column n -> hidden g*16+n
//   k < 100        : w_hh[q*100 + hidx][k]
//   k in [128,138) : pregates[k-128] = emb[tok]·w_ih_row + b_ih + b_hh   (one-hot K slots)
//   else 0
// WoutPack[n][k] (appended): n = token; k<100: w_out[n][k]; k==144: b_out[n]; else 0.
__global__ void prep_kernel(const float* __restrict__ emb,
                            const float* __restrict__ w_ih,
                            const float* __restrict__ w_hh,
                            const float* __restrict__ b_ih,
                            const float* __restrict__ b_hh,
                            const float* __restrict__ w_out,
                            const float* __restrict__ b_out,
                            _Float16* __restrict__ Wp,
                            float* __restrict__ out_tail, int draws) {
    int jid = blockIdx.x * 64 + threadIdx.x;
    if (jid < NTILE * 16) {
        int t = jid >> 4, n = jid & 15;
        int g = t >> 2, q = t & 3;
        int hidx = g * 16 + n;
        bool valid = hidx < EDIM;
        int rw = q * EDIM + hidx;
        _Float16* dst = Wp + (size_t)(t * 16 + n) * KP;
        for (int k = 0; k < KP; ++k) {
            float v = 0.f;
            if (valid) {
                if (k < EDIM) {
                    v = w_hh[rw * EDIM + k];
                } else if (k >= 128 && k < 128 + TTOK) {
                    int tk = k - 128;
                    float s = b_ih[rw] + b_hh[rw];
                    for (int e = 0; e < EDIM; ++e)
                        s += emb[tk * EDIM + e] * w_ih[rw * EDIM + e];
                    v = s;
                }
            }
            dst[k] = (_Float16)v;
        }
    } else if (jid < NTILE * 16 + 16) {
        int n = jid - NTILE * 16;
        _Float16* dst = Wp + (size_t)(NTILE * 16 + n) * KP;
        for (int k = 0; k < KP; ++k) {
            float v = 0.f;
            if (n < TTOK) {
                if (k < EDIM)      v = w_out[n * EDIM + k];
                else if (k == 144) v = b_out[n];
            }
            dst[k] = (_Float16)v;
        }
        if (n == 0) {            // zero accumulators every launch (graph replay safe)
            out_tail[0] = 0.f;                 // entropy
            out_tail[1] = 0.f;                 // matches
            out_tail[2] = (float)draws;        // stochastic draws
        }
    }
}

// ---------------- main fused 6-step LSTM decoder ----------------
// 256 threads = 8 waves; each wave owns 16 batch rows for the whole recurrence.
// Weights live in LDS (copied once per block via async-to-LDS). Steady state:
// double-buffered ds_load B fragments feeding 4 interleaved v_wmma chains.
extern __shared__ __attribute__((aligned(32))) char smem[];   // dynamic LDS @ offset 0

__global__ __launch_bounds__(256)
void lstm_decode_kernel(const float* __restrict__ h_t,
                        const _Float16* __restrict__ Wp,
                        float* __restrict__ out, long tail_off) {
    _Float16* Wl = (_Float16*)smem;                    // [WELEMS]   weights (B-layout)
    _Float16* Hl = (_Float16*)(smem + WBYTES);         // [8][16*KP] h staging (A-layout)
    int*      Tk = (int*)(smem + WBYTES + HBYTES);     // [8][16]    sampled tokens

    const int tid  = threadIdx.x;
    const int lane = tid & 31;
    const int w    = tid >> 5;
    const int half = lane >> 4;      // 0: rows r, K 0..15 ; 1: rows r+8, K 16..31
    const int ln   = lane & 15;      // M (A rows) / N (B cols, D cols)
    const long rowbase = ((long)blockIdx.x * 8 + w) * 16;
    const int a_koff = half << 4;
    _Float16* Hw = Hl + w * (16 * KP);
    const unsigned b_base = (unsigned)ln * KP + a_koff;   // lane part of B fragment addr

    // ---- async copy packed weights global -> LDS (CDNA5 async-to-LDS path)
    for (unsigned ofs = (unsigned)tid * 16u; ofs < (unsigned)WBYTES; ofs += 256u * 16u) {
        asm volatile("global_load_async_to_lds_b128 %0, %1, %2"
                     :: "v"(ofs), "v"(ofs), "s"(Wp) : "memory");
    }

    // ---- meanwhile init wave-private h tile:
    //      cols 0..99 = f16(h_t), one-hot(tok=0) col 128, const-1 col 144
    for (int idx = lane; idx < 16 * EDIM; idx += 32) {
        int m = idx / EDIM, k = idx % EDIM;
        Hw[m * KP + k] = (_Float16)h_t[(rowbase + m) * EDIM + k];
    }
    for (int idx = lane; idx < 16 * (KP - EDIM); idx += 32) {
        int m = idx / (KP - EDIM), k = EDIM + idx % (KP - EDIM);
        Hw[m * KP + k] = (_Float16)((k == 128 || k == 144) ? 1.f : 0.f);
    }

    asm volatile("s_wait_asynccnt 0x0" ::: "memory");
    __syncthreads();

    float c_state[NGRP][8];
    #pragma unroll
    for (int g = 0; g < NGRP; ++g)
        #pragma unroll
        for (int r = 0; r < 8; ++r) c_state[g][r] = 0.f;

    float ent_local = 0.f;
    int match_local = 0;

    #pragma unroll 1
    for (int step = 0; step < LSTEP; ++step) {
        // A fragments (old h), kept in VGPRs across all gate tiles
        v16h A[5];
        #pragma unroll
        for (int kt = 0; kt < 5; ++kt)
            A[kt] = *(const v16h*)&Hw[ln * KP + kt * 32 + a_koff];

        // ---- gates + LSTM cell, one 16-hidden-unit group at a time.
        //      4 gate chains interleaved per K-tile, B fragments double-buffered.
        #pragma unroll
        for (int g = 0; g < NGRP; ++g) {
            v8f acc[4] = {{}, {}, {}, {}};
            v16h Bc[4], Bn[4];
            #pragma unroll
            for (int q = 0; q < 4; ++q)
                Bc[q] = *(const v16h*)&Wl[(unsigned)((g * 4 + q) * 16) * KP + b_base];
            #pragma unroll
            for (int kt = 0; kt < 5; ++kt) {
                if (kt < 4) {
                    #pragma unroll
                    for (int q = 0; q < 4; ++q)
                        Bn[q] = *(const v16h*)&Wl[(unsigned)((g * 4 + q) * 16) * KP
                                                  + (kt + 1) * 32 + b_base];
                }
                #pragma unroll
                for (int q = 0; q < 4; ++q)
                    acc[q] = __builtin_amdgcn_wmma_f32_16x16x32_f16(
                                false, A[kt], false, Bc[q], (short)0, acc[q], false, false);
                if (kt < 4) {
                    #pragma unroll
                    for (int q = 0; q < 4; ++q) Bc[q] = Bn[q];
                }
            }
            #pragma unroll
            for (int r = 0; r < 8; ++r) {
                float iv = 1.f / (1.f + __expf(-acc[0][r]));
                float fv = 1.f / (1.f + __expf(-acc[1][r]));
                float gv = 1.f - 2.f / (__expf(2.f * acc[2][r]) + 1.f);   // tanh
                float ov = 1.f / (1.f + __expf(-acc[3][r]));
                float c  = fv * c_state[g][r] + iv * gv;
                c_state[g][r] = c;
                float hv = ov * (1.f - 2.f / (__expf(2.f * c) + 1.f));
                Hw[(r + half * 8) * KP + g * 16 + ln] = (_Float16)hv;
            }
        }

        // ---- logits = h_new · w_outᵀ (+ b_out via const-1 column), double-buffered B
        #pragma unroll
        for (int kt = 0; kt < 5; ++kt)
            A[kt] = *(const v16h*)&Hw[ln * KP + kt * 32 + a_koff];
        v8f lg = {};
        {
            v16h b0 = *(const v16h*)&Wl[(unsigned)(NTILE * 16) * KP + b_base];
            #pragma unroll
            for (int kt = 0; kt < 5; ++kt) {
                v16h b1;
                if (kt < 4)
                    b1 = *(const v16h*)&Wl[(unsigned)(NTILE * 16) * KP
                                           + (kt + 1) * 32 + b_base];
                lg = __builtin_amdgcn_wmma_f32_16x16x32_f16(
                        false, A[kt], false, b0, (short)0, lg, false, false);
                if (kt < 4) b0 = b1;
            }
        }

        // ---- softmax / entropy / argmax / categorical sample (N = lane axis)
        const bool vt = (ln < TTOK);
        #pragma unroll
        for (int r = 0; r < 8; ++r) {
            float x = vt ? lg[r] : -1e30f;
            float m = x;
            #pragma unroll
            for (int off = 1; off < 16; off <<= 1)
                m = fmaxf(m, lane_bperm(m, lane ^ off));
            float e = vt ? __expf(x - m) : 0.f;
            float s = e;
            #pragma unroll
            for (int off = 1; off < 16; off <<= 1)
                s += lane_bperm(s, lane ^ off);
            float p = e / s;
            if (vt) { float pe = p + 1e-8f; ent_local -= pe * __logf(pe); }

            unsigned bal = __builtin_amdgcn_ballot_w32(vt && (x == m));
            int amax_lo = (bal & 0xFFFFu) ? __builtin_ctz(bal & 0xFFFFu) : 0;
            int amax_hi = ((bal >> 16) & 0xFFFFu) ? __builtin_ctz((bal >> 16) & 0xFFFFu) : 0;

            // inclusive scan of e over 16 lanes -> unnormalized CDF
            float cdf = e;
            #pragma unroll
            for (int off = 1; off < 16; off <<= 1) {
                float o = lane_bperm(cdf, lane - off);
                if (ln >= off) cdf += o;
            }
            long row = rowbase + r + half * 8;
            float u = rng01((unsigned)(row * LSTEP + step) * 0x9E3779B9u + 0x85EBCA6Bu) * s;
            unsigned mk = __builtin_amdgcn_ballot_w32(vt && (cdf >= u));
            int tok_lo = (mk & 0xFFFFu) ? __builtin_ctz(mk & 0xFFFFu) : (TTOK - 1);
            int tok_hi = ((mk >> 16) & 0xFFFFu) ? __builtin_ctz((mk >> 16) & 0xFFFFu) : (TTOK - 1);
            if (lane == 0) {
                Tk[w * 16 + r]     = tok_lo;
                Tk[w * 16 + r + 8] = tok_hi;
                match_local += (amax_lo == tok_lo) + (amax_hi == tok_hi);
            }
        }

        // ---- emit tokens, refresh one-hot K columns for next step
        if (lane < 16) {
            int t = Tk[w * 16 + lane];
            out[(rowbase + lane) * LSTEP + step] = (float)t;
            #pragma unroll
            for (int k = 0; k < 16; ++k)
                Hw[lane * KP + 128 + k] = (_Float16)0.f;
            Hw[lane * KP + 128 + t] = (_Float16)1.f;
        }
    }

    // ---- global reductions
    #pragma unroll
    for (int off = 1; off < 32; off <<= 1)
        ent_local += lane_bperm(ent_local, lane ^ off);
    if (lane == 0) {
        atomicAdd(&out[tail_off],     ent_local);
        atomicAdd(&out[tail_off + 1], (float)match_local);
    }
}

// ---------------- launch ----------------
extern "C" void kernel_launch(void* const* d_in, const int* in_sizes, int n_in,
                              void* d_out, int out_size, void* d_ws, size_t ws_size,
                              hipStream_t stream) {
    const float* h_t   = (const float*)d_in[0];
    const float* emb   = (const float*)d_in[1];
    const float* w_ih  = (const float*)d_in[2];
    const float* w_hh  = (const float*)d_in[3];
    const float* b_ih  = (const float*)d_in[4];
    const float* b_hh  = (const float*)d_in[5];
    const float* w_out = (const float*)d_in[6];
    const float* b_out = (const float*)d_in[7];
    float* out = (float*)d_out;

    const int B = in_sizes[0] / EDIM;          // 262144
    const long tail = (long)B * LSTEP;

    _Float16* Wp = (_Float16*)d_ws;            // packed weights, WBYTES = 145 KiB

    prep_kernel<<<8, 64, 0, stream>>>(emb, w_ih, w_hh, b_ih, b_hh, w_out, b_out,
                                      Wp, out + tail, B * LSTEP);

    const int rows_per_block = 8 * 16;         // 8 waves x 16 rows
    lstm_decode_kernel<<<B / rows_per_block, 256, LDS_TOTAL, stream>>>(h_t, Wp, out, tail);
}
